// PhysicsInformedMHDSolver_53704271069688
// MI455X (gfx1250) — matrix operand order repaired and checked by reading, not compiled, not run on hardware
//
#include <hip/hip_runtime.h>
#include <math.h>

#define BATCH 16384
#define GRIDN 64

typedef __attribute__((ext_vector_type(16))) _Float16 v16h;
typedef __attribute__((ext_vector_type(8)))  _Float16 v8h;
typedef __attribute__((ext_vector_type(8)))  float    v8f;

// ---------------------------------------------------------------------------
// Pack f32 weight matrix W[K][N] (row-major) into WMMA B-fragment order:
// out[(((ktile*(N/16) + ntile)*32 + lane)*16 + i)] =
//     (f16) W[ktile*32 + kk + (i&7) + ((i&8)<<1)][ntile*16 + (lane&15)],  kk = (lane<16)?0:8.
// Each lane then fetches its v16h with two contiguous b128 loads.
// ---------------------------------------------------------------------------
__global__ __launch_bounds__(256)
void pack_w16_kernel(const float* __restrict__ W, _Float16* __restrict__ out,
                     int K, int N) {
    size_t idx = (size_t)blockIdx.x * blockDim.x + threadIdx.x;
    size_t total = (size_t)K * N;
    if (idx >= total) return;
    int i    = (int)(idx & 15);
    int lane = (int)((idx >> 4) & 31);
    size_t t = idx >> 9;
    int ntiles = N / 16;
    int ntile = (int)(t % ntiles);
    int ktile = (int)(t / ntiles);
    int kk = (lane < 16) ? 0 : 8;
    int k = ktile * 32 + kk + (i & 7) + ((i & 8) << 1);
    int n = ntile * 16 + (lane & 15);
    out[idx] = (_Float16)W[(size_t)k * N + n];
}

__device__ __forceinline__ float gelu_exact(float x) {
    return 0.5f * x * (1.0f + erff(x * 0.70710678118654752f));
}

// Direct global A-fragment load: per CDNA5 16x32 f16 A layout, lane (m=lane&15,
// base=(lane<16)?0:8) needs halves [base..base+7] and [16+base..16+base+7] of
// row m -- two aligned 16B chunks, no LDS staging / barriers required.
__device__ __forceinline__ v16h load_a_frag(const _Float16* __restrict__ A,
                                            size_t row_base, int K,
                                            int k0, int lane) {
    const int m = lane & 15;
    const int base = (lane < 16) ? 0 : 8;
    const _Float16* p = A + (row_base + m) * K + k0 + base;
    v8h lo = *(const v8h*)p;
    v8h hi = *(const v8h*)(p + 16);
    return __builtin_shufflevector(lo, hi, 0, 1, 2, 3, 4, 5, 6, 7,
                                   8, 9, 10, 11, 12, 13, 14, 15);
}

// ---------------------------------------------------------------------------
// Encoder: h0 = LN(GELU(x @ W_enc + b)), x:[B,8], W:[8,256]. One block per row.
// ---------------------------------------------------------------------------
__global__ __launch_bounds__(256)
void encoder_kernel(const float* __restrict__ x, const float* __restrict__ W,
                    const float* __restrict__ bias, const float* __restrict__ gamma,
                    const float* __restrict__ beta, _Float16* __restrict__ h0) {
    __shared__ float xs[8];
    __shared__ float red[16];
    const int b = blockIdx.x, t = threadIdx.x;
    if (t < 8) xs[t] = x[(size_t)b * 8 + t];
    __syncthreads();
    float acc = bias[t];
#pragma unroll
    for (int k = 0; k < 8; k++) acc += xs[k] * W[k * 256 + t];
    float g = gelu_exact(acc);
    float s = g, s2 = g * g;
#pragma unroll
    for (int off = 16; off > 0; off >>= 1) {
        s  += __shfl_xor(s, off, 32);
        s2 += __shfl_xor(s2, off, 32);
    }
    const int wave = t >> 5, lane = t & 31;
    if (lane == 0) { red[wave] = s; red[8 + wave] = s2; }
    __syncthreads();
    if (t == 0) {
        float S = 0.f, S2 = 0.f;
        for (int w = 0; w < 8; w++) { S += red[w]; S2 += red[8 + w]; }
        float mu = S / 256.f;
        red[0] = mu;
        red[1] = S2 / 256.f - mu * mu;
    }
    __syncthreads();
    float inv = rsqrtf(red[1] + 1e-5f);
    h0[(size_t)b * 256 + t] = (_Float16)((g - red[0]) * inv * gamma[t] + beta[t]);
}

// ---------------------------------------------------------------------------
// GEMM + GELU + LayerNorm.  Workgroup = 16 rows x full N (LN needs the row).
// 8 waves, each owns NT = N/128 contiguous 16x16 tiles.  Barrier-free K-loop:
// A fragments direct from global (2x b128), B fragments from packed weights.
// ---------------------------------------------------------------------------
template <int K, int N>
__global__ __launch_bounds__(256)
void gemm_gelu_ln_kernel(const _Float16* __restrict__ Aact,   // [B,K] f16
                         const _Float16* __restrict__ Wp,     // packed f16
                         const float* __restrict__ bias,
                         const float* __restrict__ gamma,
                         const float* __restrict__ beta,
                         _Float16* __restrict__ Hout) {       // [B,N] f16
    constexpr int NT = N / 128;
    __shared__ float hbuf[16 * N];
    const int tid = threadIdx.x;
    const int wave = tid >> 5, lane = tid & 31;
    const size_t m0 = (size_t)blockIdx.x * 16;
    const int ntile0 = wave * NT;

    v8f acc[NT];
    v8f zero = {};
#pragma unroll
    for (int t = 0; t < NT; t++) acc[t] = zero;

#pragma unroll
    for (int ks = 0; ks < K / 32; ks++) {
        const v16h a = load_a_frag(Aact, m0, K, ks * 32, lane);
#pragma unroll
        for (int t = 0; t < NT; t++) {
            const v16h b = *(const v16h*)(Wp +
                ((((size_t)ks * (N / 16) + ntile0 + t) * 32 + lane) << 4));
            acc[t] = __builtin_amdgcn_wmma_f32_16x16x32_f16(
                false, a, false, b, (short)0, acc[t], false, false);
        }
    }
    // bias + GELU into LDS row buffer
#pragma unroll
    for (int t = 0; t < NT; t++) {
        int n = (ntile0 + t) * 16 + (lane & 15);
        int mbase = (lane < 16) ? 0 : 8;
#pragma unroll
        for (int r = 0; r < 8; r++) {
            float v = acc[t][r] + bias[n];
            hbuf[(mbase + r) * N + n] = gelu_exact(v);
        }
    }
    __syncthreads();
    // LayerNorm: 8 waves handle 16 rows (2 per wave)
#pragma unroll
    for (int rr = 0; rr < 2; rr++) {
        int row = wave * 2 + rr;
        float s = 0.f, s2 = 0.f;
        for (int n = lane; n < N; n += 32) {
            float v = hbuf[row * N + n];
            s += v; s2 += v * v;
        }
#pragma unroll
        for (int off = 16; off > 0; off >>= 1) {
            s  += __shfl_xor(s, off, 32);
            s2 += __shfl_xor(s2, off, 32);
        }
        float mu = s / (float)N;
        float inv = rsqrtf(s2 / (float)N - mu * mu + 1e-5f);
        for (int n = lane; n < N; n += 32) {
            float v = (hbuf[row * N + n] - mu) * inv * gamma[n] + beta[n];
            Hout[(m0 + row) * N + n] = (_Float16)v;
        }
    }
}

// ---------------------------------------------------------------------------
// Final GEMM: [B,256] x [256,4096] + bias -> tanh -> sigmoid -> d_out flux
// (pre-cummax).  Block computes 16x512; each wave owns 4 tiles so the A
// fragment is amortized over 4 WMMAs.  No LDS, no barriers.
// ---------------------------------------------------------------------------
__global__ __launch_bounds__(256)
void gemm_flux_kernel(const _Float16* __restrict__ Aact,   // [B,256]
                      const _Float16* __restrict__ Wp,     // packed 256x4096
                      const float* __restrict__ bias,
                      float* __restrict__ flux) {          // [B,4096]
    constexpr int K = 256, N = 4096, NT = 4;
    const int tid = threadIdx.x, wave = tid >> 5, lane = tid & 31;
    const size_t m0 = (size_t)blockIdx.x * 16;
    const int ntile0 = blockIdx.y * 32 + wave * NT;

    v8f acc[NT];
    v8f zero = {};
#pragma unroll
    for (int t = 0; t < NT; t++) acc[t] = zero;

#pragma unroll
    for (int ks = 0; ks < K / 32; ks++) {
        const v16h a = load_a_frag(Aact, m0, K, ks * 32, lane);
#pragma unroll
        for (int t = 0; t < NT; t++) {
            const v16h b = *(const v16h*)(Wp +
                ((((size_t)ks * (N / 16) + ntile0 + t) * 32 + lane) << 4));
            acc[t] = __builtin_amdgcn_wmma_f32_16x16x32_f16(
                false, a, false, b, (short)0, acc[t], false, false);
        }
    }
#pragma unroll
    for (int t = 0; t < NT; t++) {
        int n = (ntile0 + t) * 16 + (lane & 15);
        int mbase = (lane < 16) ? 0 : 8;
#pragma unroll
        for (int r = 0; r < 8; r++) {
            float x = acc[t][r] + bias[n];
            float th = tanhf(x);
            float s = 1.0f / (1.0f + expf(-th));
            flux[(m0 + mbase + r) * (size_t)N + n] = s;
        }
    }
}

// ---------------------------------------------------------------------------
// Per-batch physics: cummax along grid columns, Laplacian residual,
// flux sum -> pressure, per-batch partials (deterministic: no float atomics).
// One block (64 threads) per batch element; grid in padded LDS (stride 65).
// ---------------------------------------------------------------------------
__global__ __launch_bounds__(64)
void physics_kernel(const float* __restrict__ plasma,   // [B,8]
                    float* __restrict__ flux,           // [B,4096] in/out
                    float* __restrict__ pressure,       // [B]
                    float* __restrict__ part_gs,        // [B]
                    float* __restrict__ part_cc) {      // [B]
    __shared__ float g[GRIDN * 65];
    __shared__ float red[4];
    const int b = blockIdx.x, t = threadIdx.x;
    const int wave = t >> 5, lane = t & 31;
    float* base = flux + (size_t)b * (GRIDN * GRIDN);
    for (int i = t; i < GRIDN * GRIDN; i += 64)
        g[(i >> 6) * 65 + (i & 63)] = base[i];
    __syncthreads();
    // running max along columns (axis=2); thread t owns grid row t
    {
        float run = g[t * 65];
        for (int j = 1; j < GRIDN; j++) {
            float v = g[t * 65 + j];
            run = fmaxf(run, v);
            g[t * 65 + j] = run;
        }
    }
    __syncthreads();
    // row sums + Laplacian residual for row t
    const float src = plasma[(size_t)b * 8];
    float rs = 0.f, gs = 0.f;
    for (int j = 0; j < GRIDN; j++) {
        float c = g[t * 65 + j];
        rs += c;
        float lap = 0.f;
        if (t >= 1 && t <= GRIDN - 2 && j >= 1 && j <= GRIDN - 2) {
            lap = g[(t + 1) * 65 + j] + g[(t - 1) * 65 + j]
                + g[t * 65 + j + 1] + g[t * 65 + j - 1] - 4.0f * c;
        }
        float r = lap - src;
        gs += r * r;
    }
#pragma unroll
    for (int off = 16; off > 0; off >>= 1) {
        rs += __shfl_xor(rs, off, 32);
        gs += __shfl_xor(gs, off, 32);
    }
    if (lane == 0) { red[wave] = rs; red[2 + wave] = gs; }
    // write cummax'd flux back
    for (int i = t; i < GRIDN * GRIDN; i += 64)
        base[i] = g[(i >> 6) * 65 + (i & 63)];
    __syncthreads();
    if (t == 0) {
        float S  = red[0] + red[1];
        float GS = red[2] + red[3];
        pressure[b] = expf(-2.0f * S / (float)(GRIDN * GRIDN));
        float d = src - S;
        part_cc[b] = d * d;
        part_gs[b] = GS;
    }
}

__global__ __launch_bounds__(256)
void qfill_kernel(float* __restrict__ q) {
    int idx = blockIdx.x * blockDim.x + threadIdx.x;
    if (idx < BATCH * GRIDN) {
        int j = idx & (GRIDN - 1);
        q[idx] = 0.8f + (2.7f / 63.0f) * (float)j;
    }
}

// Deterministic fixed-order reduction of per-batch partials -> scalar loss.
// q_profile min is the linspace start 0.8 for every row => stability = relu(0.3).
__global__ __launch_bounds__(256)
void finalize_kernel(const float* __restrict__ part_gs,
                     const float* __restrict__ part_cc,
                     const float* __restrict__ pw,
                     float* __restrict__ loss_out) {
    __shared__ float r1[256], r2[256];
    const int t = threadIdx.x;
    float s1 = 0.f, s2 = 0.f;
    for (int i = t; i < BATCH; i += 256) { s1 += part_gs[i]; s2 += part_cc[i]; }
    r1[t] = s1; r2[t] = s2;
    __syncthreads();
    for (int off = 128; off > 0; off >>= 1) {
        if (t < off) { r1[t] += r1[t + off]; r2[t] += r2[t + off]; }
        __syncthreads();
    }
    if (t == 0) {
        float gs = r1[0] / ((float)BATCH * (float)(GRIDN * GRIDN));
        float cc = r2[0] / (float)BATCH;
        float stability = fmaxf(1.1f - 0.8f, 0.0f);
        loss_out[0] = pw[0] * gs + pw[1] * cc + pw[2] * stability;
    }
}

// ---------------------------------------------------------------------------
extern "C" void kernel_launch(void* const* d_in, const int* in_sizes, int n_in,
                              void* d_out, int out_size, void* d_ws, size_t ws_size,
                              hipStream_t stream) {
    const float* plasma = (const float*)d_in[0];
    const float* W_enc  = (const float*)d_in[1];
    const float* b_enc  = (const float*)d_in[2];
    const float* g_enc  = (const float*)d_in[3];
    const float* be_enc = (const float*)d_in[4];
    const float* W1     = (const float*)d_in[5];
    const float* b1     = (const float*)d_in[6];
    const float* g1     = (const float*)d_in[7];
    const float* be1    = (const float*)d_in[8];
    const float* W2     = (const float*)d_in[9];
    const float* b2     = (const float*)d_in[10];
    const float* g2     = (const float*)d_in[11];
    const float* be2    = (const float*)d_in[12];
    const float* Wf     = (const float*)d_in[13];
    const float* bf     = (const float*)d_in[14];
    const float* pw     = (const float*)d_in[15];

    float* out      = (float*)d_out;
    float* flux     = out;                                   // [B,64,64]
    float* qprof    = flux + (size_t)BATCH * GRIDN * GRIDN;  // [B,64]
    float* pressure = qprof + (size_t)BATCH * GRIDN;         // [B]
    float* loss     = pressure + BATCH;                      // [1]

    size_t off = 0;
    auto carve = [&](size_t bytes) -> void* {
        void* p = (char*)d_ws + off;
        off += (bytes + 255) & ~(size_t)255;
        return p;
    };
    _Float16* h0  = (_Float16*)carve((size_t)BATCH * 256 * 2);
    _Float16* h1  = (_Float16*)carve((size_t)BATCH * 512 * 2);
    _Float16* h2  = (_Float16*)carve((size_t)BATCH * 256 * 2);
    _Float16* W1p = (_Float16*)carve((size_t)256 * 512 * 2);
    _Float16* W2p = (_Float16*)carve((size_t)512 * 256 * 2);
    _Float16* Wfp = (_Float16*)carve((size_t)256 * 4096 * 2);
    float* part_gs = (float*)carve((size_t)BATCH * 4);
    float* part_cc = (float*)carve((size_t)BATCH * 4);
    (void)ws_size; (void)in_sizes; (void)n_in; (void)out_size;

    // weight packing (f32 -> f16 WMMA B-fragment order)
    pack_w16_kernel<<<(256 * 512 + 255) / 256, 256, 0, stream>>>(W1, W1p, 256, 512);
    pack_w16_kernel<<<(512 * 256 + 255) / 256, 256, 0, stream>>>(W2, W2p, 512, 256);
    pack_w16_kernel<<<(256 * 4096 + 255) / 256, 256, 0, stream>>>(Wf, Wfp, 256, 4096);

    // encoder
    encoder_kernel<<<BATCH, 256, 0, stream>>>(plasma, W_enc, b_enc, g_enc, be_enc, h0);

    // hidden layers (WMMA f16, fused GELU + LN), barrier-free K loops
    gemm_gelu_ln_kernel<256, 512><<<BATCH / 16, 256, 0, stream>>>(h0, W1p, b1, g1, be1, h1);
    gemm_gelu_ln_kernel<512, 256><<<BATCH / 16, 256, 0, stream>>>(h1, W2p, b2, g2, be2, h2);

    // flux head (WMMA f16, fused tanh+sigmoid) -> d_out
    dim3 gf(BATCH / 16, 4096 / 512);
    gemm_flux_kernel<<<gf, 256, 0, stream>>>(h2, Wfp, bf, flux);

    // cummax + physics partials (in-place on d_out flux)
    physics_kernel<<<BATCH, 64, 0, stream>>>(plasma, flux, pressure, part_gs, part_cc);

    // q profile + scalar loss
    qfill_kernel<<<(BATCH * GRIDN + 255) / 256, 256, 0, stream>>>(qprof);
    finalize_kernel<<<1, 256, 0, stream>>>(part_gs, part_cc, pw, loss);
}
